// BertAttention_5514738008565
// MI455X (gfx1250) — compile-verified, matrix operand order checked by reading, not run on hardware
//
#include <hip/hip_runtime.h>
#include <hip/hip_bf16.h>

#define Bb 4
#define Ss 2048
#define Hh 1024
#define NH 16
#define DH 64
#define LN_EPS 1e-5f

typedef unsigned short u16;
typedef __bf16  bf16x16 __attribute__((ext_vector_type(16)));
typedef u16     u16x16  __attribute__((ext_vector_type(16)));
typedef float   floatx8 __attribute__((ext_vector_type(8)));

static __device__ __forceinline__ u16 f2bf(float f) {
    unsigned int u = __float_as_uint(f);
    u += 0x7FFFu + ((u >> 16) & 1u);   // round-to-nearest-even
    return (u16)(u >> 16);
}

static __device__ __forceinline__ bf16x16 ld_frag(const u16* p) {
    u16x16 u = *(const u16x16*)p;
    return __builtin_bit_cast(bf16x16, u);
}

static __device__ __forceinline__ floatx8 wmma_bf16(bf16x16 a, bf16x16 b, floatx8 c) {
    return __builtin_amdgcn_wmma_f32_16x16x32_bf16(false, a, false, b, (short)0, c, false, false);
}

// ---------------------------------------------------------------- fp32 -> bf16
__global__ void k_f2bf(const float* __restrict__ in, u16* __restrict__ out, int n) {
    int i = blockIdx.x * blockDim.x + threadIdx.x;
    if (i < n) out[i] = f2bf(in[i]);
}

// ---------------------------------------------------------------- V -> V^T per head
// V: [B, S, H] bf16  ->  Vt: [B, NH, DH, S] bf16
__global__ void k_transpose_v(const u16* __restrict__ V, u16* __restrict__ Vt) {
    int idx = blockIdx.x * blockDim.x + threadIdx.x;
    if (idx >= Bb * Ss * Hh) return;
    int hd = idx % Hh;
    int s  = (idx / Hh) % Ss;
    int b  = idx / (Ss * Hh);
    int h  = hd >> 6;
    int d  = hd & 63;
    Vt[((size_t)(b * NH + h) * DH + d) * Ss + s] = V[idx];
}

// ---------------------------------------------------------------- WMMA GEMM
// Y[M,N] = A[M,K](bf16) @ W[N,K]^T(bf16) + bias
// MODE 0: bf16 output.  MODE 1: fp32 output + residual add.
// Block: 256 threads (8 waves). Block tile 64x256; wave tile 32x64.
// Per K-step: 6 fragment loads feed 8 WMMAs (B-frags amortized over 2 M-tiles).
template <int MODE>
__global__ void k_gemm(const u16* __restrict__ A, const u16* __restrict__ W,
                       const float* __restrict__ bias, const float* __restrict__ resid,
                       u16* __restrict__ obf, float* __restrict__ of,
                       int M, int N, int K) {
    int tid = threadIdx.x;
    int w = tid >> 5, lane = tid & 31, lr = lane & 15, lh = lane >> 4;
    int m0 = (blockIdx.x * 2 + (w >> 2)) * 32;
    int n0 = (blockIdx.y * 4 + (w & 3)) * 64;

    floatx8 c[2][4];
#pragma unroll
    for (int g = 0; g < 2; ++g)
#pragma unroll
        for (int j = 0; j < 4; ++j) c[g][j] = floatx8{};

    const u16* ap0 = A + (size_t)(m0 +  0 + lr) * K + lh * 16;
    const u16* ap1 = A + (size_t)(m0 + 16 + lr) * K + lh * 16;
    const u16* wp0 = W + (size_t)(n0 +  0 + lr) * K + lh * 16;
    const u16* wp1 = W + (size_t)(n0 + 16 + lr) * K + lh * 16;
    const u16* wp2 = W + (size_t)(n0 + 32 + lr) * K + lh * 16;
    const u16* wp3 = W + (size_t)(n0 + 48 + lr) * K + lh * 16;

    for (int k0 = 0; k0 < K; k0 += 32) {
        bf16x16 a0 = ld_frag(ap0 + k0);
        bf16x16 a1 = ld_frag(ap1 + k0);
        __builtin_prefetch(ap0 + k0 + 256, 0, 1);  // global_prefetch_b8, ~8 iters ahead
        __builtin_prefetch(ap1 + k0 + 256, 0, 1);
        bf16x16 b0 = ld_frag(wp0 + k0);
        c[0][0] = wmma_bf16(a0, b0, c[0][0]);
        c[1][0] = wmma_bf16(a1, b0, c[1][0]);
        bf16x16 b1 = ld_frag(wp1 + k0);
        c[0][1] = wmma_bf16(a0, b1, c[0][1]);
        c[1][1] = wmma_bf16(a1, b1, c[1][1]);
        bf16x16 b2 = ld_frag(wp2 + k0);
        c[0][2] = wmma_bf16(a0, b2, c[0][2]);
        c[1][2] = wmma_bf16(a1, b2, c[1][2]);
        bf16x16 b3 = ld_frag(wp3 + k0);
        c[0][3] = wmma_bf16(a0, b3, c[0][3]);
        c[1][3] = wmma_bf16(a1, b3, c[1][3]);
    }

#pragma unroll
    for (int g = 0; g < 2; ++g) {
#pragma unroll
        for (int j = 0; j < 4; ++j) {
            int col = n0 + j * 16 + lr;
            float bv = bias[col];
#pragma unroll
            for (int i = 0; i < 8; ++i) {
                int row = m0 + g * 16 + i + lh * 8;
                size_t off = (size_t)row * N + col;
                float v = c[g][j][i] + bv;
                if (MODE == 0) {
                    obf[off] = f2bf(v);
                } else {
                    of[off] = v + resid[off];
                }
            }
        }
    }
}

// ---------------------------------------------------------------- fused attention
// One wave per (bh, 32-query tile). Streams K/V in 32-key blocks:
//   scores = Q K^T * scale + mask  -> written to d_out (fp32, once)
//   online softmax + context accumulation (WMMA vs V^T)
// K-frags and V-frags are each reused by two query groups: 16 WMMA per 8 loads.
__global__ void __launch_bounds__(32)
k_attn(const u16* __restrict__ Qbf, const u16* __restrict__ Kbf,
       const u16* __restrict__ Vt, const float* __restrict__ mask,
       float* __restrict__ scores, u16* __restrict__ ctx) {
    const int lane = threadIdx.x;
    const int lr = lane & 15, lh = lane >> 4;
    const int bh = blockIdx.y;
    const int b = bh / NH, h = bh % NH;
    const int q0 = blockIdx.x * 32;
    const float scale = 0.125f;  // 1/sqrt(64)

    __shared__ u16 pbuf[2][16 * 32];

    // Q fragments: 2 groups of 16 rows x 64 dh, resident for the whole sweep
    bf16x16 a[2][2];
#pragma unroll
    for (int g = 0; g < 2; ++g) {
        const u16* qbase = Qbf + ((size_t)(b * Ss + q0 + g * 16 + lr)) * Hh + h * DH;
        a[g][0] = ld_frag(qbase + lh * 16);
        a[g][1] = ld_frag(qbase + 32 + lh * 16);
    }

    floatx8 o[2][4];
    float mrow[2][8], lrow[2][8];
#pragma unroll
    for (int g = 0; g < 2; ++g) {
#pragma unroll
        for (int j = 0; j < 4; ++j) o[g][j] = floatx8{};
#pragma unroll
        for (int i = 0; i < 8; ++i) { mrow[g][i] = -1e30f; lrow[g][i] = 0.0f; }
    }

    float* sc_base = scores + (size_t)bh * Ss * Ss;

    for (int kb = 0; kb < Ss / 32; ++kb) {
        float sreg[2][2][8];  // [qgroup][keytile][i]
#pragma unroll
        for (int t = 0; t < 2; ++t) {
            int key16 = kb * 32 + t * 16;
            const u16* kbase = Kbf + ((size_t)(b * Ss + key16 + lr)) * Hh + h * DH;
            bf16x16 kb0 = ld_frag(kbase + lh * 16);
            bf16x16 kb1 = ld_frag(kbase + 32 + lh * 16);
            float mv = -10000.0f * mask[b * Ss + key16 + lr];  // col = lr
#pragma unroll
            for (int g = 0; g < 2; ++g) {
                floatx8 s = {};
                s = wmma_bf16(a[g][0], kb0, s);
                s = wmma_bf16(a[g][1], kb1, s);
#pragma unroll
                for (int i = 0; i < 8; ++i) {
                    float v = s[i] * scale + mv;
                    sreg[g][t][i] = v;
                    int q = q0 + g * 16 + i + lh * 8;
                    sc_base[(size_t)q * Ss + key16 + lr] = v;  // masked score -> output 2
                }
            }
        }
        // online softmax over these 32 keys (rows live across a 16-lane half)
#pragma unroll
        for (int g = 0; g < 2; ++g) {
#pragma unroll
            for (int i = 0; i < 8; ++i) {
                float rmax = fmaxf(sreg[g][0][i], sreg[g][1][i]);
                rmax = fmaxf(rmax, __shfl_xor(rmax, 1, 32));
                rmax = fmaxf(rmax, __shfl_xor(rmax, 2, 32));
                rmax = fmaxf(rmax, __shfl_xor(rmax, 4, 32));
                rmax = fmaxf(rmax, __shfl_xor(rmax, 8, 32));
                float mnew = fmaxf(mrow[g][i], rmax);
                float fs = __expf(mrow[g][i] - mnew);
                float p0 = __expf(sreg[g][0][i] - mnew);
                float p1 = __expf(sreg[g][1][i] - mnew);
                float rsum = p0 + p1;
                rsum += __shfl_xor(rsum, 1, 32);
                rsum += __shfl_xor(rsum, 2, 32);
                rsum += __shfl_xor(rsum, 4, 32);
                rsum += __shfl_xor(rsum, 8, 32);
                lrow[g][i] = lrow[g][i] * fs + rsum;
                mrow[g][i] = mnew;
                o[g][0][i] *= fs; o[g][1][i] *= fs; o[g][2][i] *= fs; o[g][3][i] *= fs;
                int r = i + lh * 8;
                pbuf[g][r * 32 + lr]      = f2bf(p0);  // C-layout -> row-major P in LDS
                pbuf[g][r * 32 + 16 + lr] = f2bf(p1);
            }
        }
        __syncthreads();
        // read back as A-fragments (16 q-rows x 32 keys), contiguous ds loads
        bf16x16 pf0 = ld_frag(&pbuf[0][lr * 32 + lh * 16]);
        bf16x16 pf1 = ld_frag(&pbuf[1][lr * 32 + lh * 16]);
        __syncthreads();

        const u16* vb = Vt + ((size_t)bh * DH + lr) * Ss + kb * 32 + lh * 16;
#pragma unroll
        for (int j = 0; j < 4; ++j) {
            bf16x16 vf = ld_frag(vb + (size_t)j * 16 * Ss);
            o[0][j] = wmma_bf16(pf0, vf, o[0][j]);
            o[1][j] = wmma_bf16(pf1, vf, o[1][j]);
        }
    }

    // normalize and store context (bf16, [B,S,H] layout)
#pragma unroll
    for (int g = 0; g < 2; ++g) {
#pragma unroll
        for (int i = 0; i < 8; ++i) {
            float inv = 1.0f / lrow[g][i];
            int q = q0 + g * 16 + i + lh * 8;
            size_t rb = (size_t)(b * Ss + q) * Hh + h * DH;
            ctx[rb +  0 + lr] = f2bf(o[g][0][i] * inv);
            ctx[rb + 16 + lr] = f2bf(o[g][1][i] * inv);
            ctx[rb + 32 + lr] = f2bf(o[g][2][i] * inv);
            ctx[rb + 48 + lr] = f2bf(o[g][3][i] * inv);
        }
    }
}

// ---------------------------------------------------------------- layernorm
__global__ void k_layernorm(const float* __restrict__ x, const float* __restrict__ gamma,
                            const float* __restrict__ beta, float* __restrict__ out) {
    int row = blockIdx.x;
    int tid = threadIdx.x;
    const float* xr = x + (size_t)row * Hh;
    float s = 0.f, sq = 0.f;
    for (int c = tid; c < Hh; c += 256) { float v = xr[c]; s += v; sq += v * v; }
#pragma unroll
    for (int msk = 16; msk >= 1; msk >>= 1) {
        s  += __shfl_xor(s,  msk, 32);
        sq += __shfl_xor(sq, msk, 32);
    }
    __shared__ float shs[8], shq[8];
    int w = tid >> 5, lane = tid & 31;
    if (lane == 0) { shs[w] = s; shq[w] = sq; }
    __syncthreads();
    float ts = 0.f, tq = 0.f;
#pragma unroll
    for (int k = 0; k < 8; ++k) { ts += shs[k]; tq += shq[k]; }
    float mu = ts * (1.0f / Hh);
    float var = tq * (1.0f / Hh) - mu * mu;
    float r = rsqrtf(var + LN_EPS);
    for (int c = tid; c < Hh; c += 256)
        out[(size_t)row * Hh + c] = (xr[c] - mu) * r * gamma[c] + beta[c];
}

// ---------------------------------------------------------------- host
extern "C" void kernel_launch(void* const* d_in, const int* in_sizes, int n_in,
                              void* d_out, int out_size, void* d_ws, size_t ws_size,
                              hipStream_t stream) {
    const float* hidden = (const float*)d_in[0];
    const float* amask  = (const float*)d_in[1];
    const float* Wq = (const float*)d_in[2];
    const float* bq = (const float*)d_in[3];
    const float* Wk = (const float*)d_in[4];
    const float* bk = (const float*)d_in[5];
    const float* Wv = (const float*)d_in[6];
    const float* bv = (const float*)d_in[7];
    const float* Wo = (const float*)d_in[8];
    const float* bo = (const float*)d_in[9];
    const float* gamma = (const float*)d_in[10];
    const float* beta  = (const float*)d_in[11];

    float* out_ln     = (float*)d_out;                       // [B,S,H]
    float* out_scores = out_ln + (size_t)Bb * Ss * Hh;       // [B,NH,S,S]

    // workspace carve-up (bf16 buffers unless noted)
    char* ws = (char*)d_ws;
    size_t off = 0;
    const size_t XSZ = (size_t)Bb * Ss * Hh;       // activation elements
    const size_t WSZ = (size_t)Hh * Hh;            // weight elements
    u16* XBF = (u16*)(ws + off); off += XSZ * 2;
    u16* WQB = (u16*)(ws + off); off += WSZ * 2;
    u16* WKB = (u16*)(ws + off); off += WSZ * 2;
    u16* WVB = (u16*)(ws + off); off += WSZ * 2;
    u16* WOB = (u16*)(ws + off); off += WSZ * 2;
    u16* QBF = (u16*)(ws + off); off += XSZ * 2;
    u16* KBF = (u16*)(ws + off); off += XSZ * 2;
    u16* VBF = (u16*)(ws + off); off += XSZ * 2;
    u16* VT  = (u16*)(ws + off); off += XSZ * 2;
    u16* CTX = (u16*)(ws + off); off += XSZ * 2;
    float* YB = (float*)(ws + off); off += XSZ * 4;  // fp32 pre-LN buffer
    (void)ws_size; (void)n_in; (void)in_sizes; (void)out_size;

    const int M = Bb * Ss;  // 8192
    const int N = Hh, K = Hh;

    // 1) pack to bf16
    k_f2bf<<<(int)((XSZ + 255) / 256), 256, 0, stream>>>(hidden, XBF, (int)XSZ);
    k_f2bf<<<(int)((WSZ + 255) / 256), 256, 0, stream>>>(Wq, WQB, (int)WSZ);
    k_f2bf<<<(int)((WSZ + 255) / 256), 256, 0, stream>>>(Wk, WKB, (int)WSZ);
    k_f2bf<<<(int)((WSZ + 255) / 256), 256, 0, stream>>>(Wv, WVB, (int)WSZ);
    k_f2bf<<<(int)((WSZ + 255) / 256), 256, 0, stream>>>(Wo, WOB, (int)WSZ);

    // 2) QKV projections (WMMA), block tile 64x256
    dim3 gg(M / 64, N / 256);
    k_gemm<0><<<gg, 256, 0, stream>>>(XBF, WQB, bq, nullptr, QBF, nullptr, M, N, K);
    k_gemm<0><<<gg, 256, 0, stream>>>(XBF, WKB, bk, nullptr, KBF, nullptr, M, N, K);
    k_gemm<0><<<gg, 256, 0, stream>>>(XBF, WVB, bv, nullptr, VBF, nullptr, M, N, K);

    // 3) V transpose per head
    k_transpose_v<<<(int)((XSZ + 255) / 256), 256, 0, stream>>>(VBF, VT);

    // 4) fused attention: scores -> d_out, softmax, context (WMMA)
    k_attn<<<dim3(Ss / 32, Bb * NH), 32, 0, stream>>>(QBF, KBF, VT, amask,
                                                      out_scores, CTX);

    // 5) output projection + residual (WMMA, fp32 out)
    k_gemm<1><<<gg, 256, 0, stream>>>(CTX, WOB, bo, hidden, nullptr, YB, M, N, K);

    // 6) layernorm -> d_out
    k_layernorm<<<M, 256, 0, stream>>>(YB, gamma, beta, out_ln);
}
